// GatedAttentionUnit_7378753815212
// MI455X (gfx1250) — matrix operand order repaired and checked by reading, not compiled, not run on hardware
//
#include <hip/hip_runtime.h>
#include <hip/hip_bf16.h>

// ---------------- problem constants ----------------
#define BB 4
#define HH 96
#define WW 96
#define CC 128
#define SEQ (HH*WW)              // 9216
#define NTOK (BB*SEQ)            // 36864
#define NWIN 16                  // B * ns * ns
#define WTOK 2304                // 48*48 tokens per window
#define WH 48
#define SH 24
#define SCORE_SCALE (1.0f/((float)CC*(float)SEQ))

typedef __attribute__((ext_vector_type(16))) _Float16 v16h;
typedef __attribute__((ext_vector_type(8)))  _Float16 v8h;
typedef __attribute__((ext_vector_type(8)))  float    v8f;

// ---------------- fast math ----------------
__device__ __forceinline__ float fast_rcp(float x) { return __builtin_amdgcn_rcpf(x); }
__device__ __forceinline__ float silu(float x) { return x * fast_rcp(1.0f + __expf(-x)); }

// ---------------- WMMA fragment helpers (CDNA5 16x16x32 f16, wave32) -------
// A (16xK tile, row-major src): lane<16 -> row=lane, K = k0 + {0..7, 16..23}
//                               lane>=16 -> row=lane-16, K = k0 + {8..15, 24..31}
// p is 16B-aligned when ld, k0 are multiples of 8 halfs.
__device__ __forceinline__ v16h load_a_frag(const _Float16* base, int ld, int k0, int lane) {
  int row  = lane & 15;
  int half = lane >> 4;
  const _Float16* p = base + row * ld + k0 + half * 8;
  v8h lo = *(const v8h*)p;
  v8h hi = *(const v8h*)(p + 16);
  return __builtin_shufflevector(lo, hi, 0,1,2,3,4,5,6,7,8,9,10,11,12,13,14,15);
}

// B (KxN tile) loaded from a [N][K] ("transposed") array: per lane one column,
// K = k0 + {0..7, 16..23} for lane<16, k0 + {8..15, 24..31} for lane>=16.
__device__ __forceinline__ v16h load_b_frag(const _Float16* baseT, int ld, int k0, int lane) {
  int col  = lane & 15;
  int koff = (lane >> 4) * 8;
  const _Float16* p = baseT + col * ld + k0 + koff;
  v8h lo = *(const v8h*)p;
  v8h hi = *(const v8h*)(p + 16);
  return __builtin_shufflevector(lo, hi, 0,1,2,3,4,5,6,7,8,9,10,11,12,13,14,15);
}

__device__ __forceinline__ v8f wmma16(v16h a, v16h b, v8f c) {
  return __builtin_amdgcn_wmma_f32_16x16x32_f16(false, a, false, b, (short)0, c, false, false);
}

// ---------------- shifted-window index maps ----------------
__device__ __forceinline__ int seq_from_win(int wi, int t) {
  int b = wi >> 2, i = (wi >> 1) & 1, j = wi & 1;
  int r = t / WH, cc = t - r * WH;
  int h0 = i * WH + r + SH;  if (h0 >= HH) h0 -= HH;
  int w0 = j * WH + cc + SH; if (w0 >= WW) w0 -= WW;
  return b * SEQ + h0 * WW + w0;
}
__device__ __forceinline__ void win_from_seq(int n, int& wi, int& t) {
  int b = n / SEQ, rem = n - b * SEQ;
  int h0 = rem / WW, w0 = rem - h0 * WW;
  int hr = h0 - SH; if (hr < 0) hr += HH;
  int wr = w0 - SH; if (wr < 0) wr += WW;
  int i = hr / WH, r = hr - i * WH;
  int j = wr / WH, cc = wr - j * WH;
  wi = b * 4 + i * 2 + j;
  t = r * WH + cc;
}

// ---------------- Kernel 1: LayerNorm -> f16 ----------------
__global__ __launch_bounds__(256) void ln_kernel(const float* __restrict__ src,
                                                 const float* __restrict__ g,
                                                 const float* __restrict__ b,
                                                 _Float16* __restrict__ out) {
  int tok  = (blockIdx.x * blockDim.x + threadIdx.x) >> 5;
  int lane = threadIdx.x & 31;
  if (tok >= NTOK) return;
  const float4 v = *(const float4*)(src + tok * CC + lane * 4);
  float s  = v.x + v.y + v.z + v.w;
  float s2 = v.x*v.x + v.y*v.y + v.z*v.z + v.w*v.w;
#pragma unroll
  for (int m = 1; m < 32; m <<= 1) { s += __shfl_xor(s, m, 32); s2 += __shfl_xor(s2, m, 32); }
  float mean = s * (1.0f / CC);
  float var  = s2 * (1.0f / CC) - mean * mean;
  float inv  = rsqrtf(var + 1e-5f);
  const float4 gg = *(const float4*)(g + lane * 4);
  const float4 bb = *(const float4*)(b + lane * 4);
  _Float16* o = out + tok * CC + lane * 4;
  o[0] = (_Float16)((v.x - mean) * inv * gg.x + bb.x);
  o[1] = (_Float16)((v.y - mean) * inv * gg.y + bb.y);
  o[2] = (_Float16)((v.z - mean) * inv * gg.z + bb.z);
  o[3] = (_Float16)((v.w - mean) * inv * gg.w + bb.w);
}

// ---------------- Kernel 2: weight transpose + f32->f16 ----------------
// w is [K][N] row-major; wT becomes [N][K]
__global__ void wcvt(const float* __restrict__ w, _Float16* __restrict__ wT, int K, int N) {
  int idx = blockIdx.x * blockDim.x + threadIdx.x;
  if (idx >= K * N) return;
  int n = idx / K, k = idx - n * K;
  wT[idx] = (_Float16)w[k * N + n];
}

// ---------------- Kernel 3: GQ/KV projections + SiLU + window scatter ------
__global__ __launch_bounds__(256) void proj_gqkv(const _Float16* __restrict__ sln,
                                                 const _Float16* __restrict__ wgqT,
                                                 const _Float16* __restrict__ wkvT,
                                                 const float* __restrict__ bgq,
                                                 const float* __restrict__ bkv,
                                                 float* __restrict__ gate,
                                                 _Float16* __restrict__ Qh,
                                                 _Float16* __restrict__ Kh,
                                                 _Float16* __restrict__ Vh) {
  int mt    = blockIdx.x;                 // 16-row tile index (0..2303)
  int wave  = threadIdx.x >> 5;
  int lane  = threadIdx.x & 31;
  int ctile = blockIdx.y * 8 + wave;      // 0..31 -> 512 output columns
  bool is_gq   = ctile < 16;
  int  colbase = (is_gq ? ctile : ctile - 16) * 16;
  const _Float16* A  = sln + mt * 16 * CC;
  const _Float16* wT = (is_gq ? wgqT : wkvT) + colbase * CC;
  const float*  bias = (is_gq ? bgq : bkv) + colbase;

  v8f acc = {0.f,0.f,0.f,0.f,0.f,0.f,0.f,0.f};
#pragma unroll
  for (int kk = 0; kk < 4; ++kk)
    acc = wmma16(load_a_frag(A, CC, kk * 32, lane),
                 load_b_frag(wT, CC, kk * 32, lane), acc);

  int col = lane & 15, hi = lane >> 4;
  float bv = bias[col];
#pragma unroll
  for (int v = 0; v < 8; ++v) {
    int m = v + hi * 8;
    int n = mt * 16 + m;
    int c = colbase + col;
    float y = silu(acc[v] + bv);
    if (is_gq) {
      if (c < CC) {
        gate[n * CC + c] = y;
      } else {
        int wi, t; win_from_seq(n, wi, t);
        Qh[(wi * WTOK + t) * CC + (c - CC)] = (_Float16)y;
      }
    } else {
      int wi, t; win_from_seq(n, wi, t);
      if (c < CC) Kh[(wi * WTOK + t) * CC + c]        = (_Float16)y;
      else        Vh[(wi * WTOK + t) * CC + (c - CC)] = (_Float16)y;
    }
  }
}

// ---------------- Kernel 4: windowed attention (flash-style, sum-only) -----
__global__ __launch_bounds__(256, 1) void attn_kernel(const _Float16* __restrict__ Qh,
                                                      const _Float16* __restrict__ Kh,
                                                      const _Float16* __restrict__ Vh,
                                                      const float* __restrict__ mask,
                                                      const float* __restrict__ gate,
                                                      _Float16* __restrict__ x2) {
  __shared__ _Float16 Ks[32 * CC];        // K chunk, [t][c]
  __shared__ _Float16 Vt[CC * 32];        // V chunk transposed, [c][t]
  __shared__ _Float16 Pt[8][16 * 32];     // per-wave exp(S) tile, row-major 16x32

  int wi   = blockIdx.x;                  // window 0..15
  int qb   = blockIdx.y;                  // q block 0..17
  int wave = threadIdx.x >> 5;
  int lane = threadIdx.x & 31;
  int q0   = qb * 128 + wave * 16;        // this wave's 16 Q rows
  int col  = lane & 15, hi = lane >> 4;
  int wtype = wi & 3;

  // hoisted per-row mask pointers (already offset by this lane's column)
  const float* mrowp[8];
#pragma unroll
  for (int v = 0; v < 8; ++v)
    mrowp[v] = mask + (size_t)wtype * WTOK * WTOK
                    + (size_t)(q0 + v + hi * 8) * WTOK + col;

  // Q fragments held in registers for the whole kernel (16x128)
  const _Float16* Qbase = Qh + (size_t)(wi * WTOK + q0) * CC;
  v16h qa[4];
#pragma unroll
  for (int kk = 0; kk < 4; ++kk) qa[kk] = load_a_frag(Qbase, CC, kk * 32, lane);

  v8f out[8];
#pragma unroll
  for (int i = 0; i < 8; ++i) out[i] = (v8f){0.f,0.f,0.f,0.f,0.f,0.f,0.f,0.f};
  float prs[8] = {0.f,0.f,0.f,0.f,0.f,0.f,0.f,0.f};

  // staging thread's fixed (t, c0) assignment
  int st  = threadIdx.x >> 3;             // token 0..31 of chunk
  int sc0 = (threadIdx.x & 7) * 16;       // 16 channels
  const _Float16* kgbase = Kh + (size_t)(wi * WTOK + st) * CC + sc0;
  const _Float16* vgbase = Vh + (size_t)(wi * WTOK + st) * CC + sc0;

  for (int kc = 0; kc < WTOK; kc += 32) {
    // cooperative stage of K (row-major) and V (transposed) into LDS
    {
      const _Float16* kg = kgbase + (size_t)kc * CC;
      const _Float16* vg = vgbase + (size_t)kc * CC;
      v8h k0v = *(const v8h*)kg;
      v8h k1v = *(const v8h*)(kg + 8);
      v8h v0v = *(const v8h*)vg;
      v8h v1v = *(const v8h*)(vg + 8);
      *(v8h*)&Ks[st * CC + sc0]     = k0v;
      *(v8h*)&Ks[st * CC + sc0 + 8] = k1v;
#pragma unroll
      for (int i = 0; i < 8; ++i) Vt[(sc0 + i) * 32 + st]     = v0v[i];
#pragma unroll
      for (int i = 0; i < 8; ++i) Vt[(sc0 + 8 + i) * 32 + st] = v1v[i];
      // prefetch next chunk toward the caches while this one computes
      if (kc + 32 < WTOK) {
        __builtin_prefetch(kg + 32 * CC, 0, 3);
        __builtin_prefetch(vg + 32 * CC, 0, 3);
      }
    }
    __syncthreads();

    // scores: two 16x16 tiles covering 32 K tokens; exp + rowsum; stash P in LDS
#pragma unroll
    for (int ct = 0; ct < 2; ++ct) {
      v8f s = {0.f,0.f,0.f,0.f,0.f,0.f,0.f,0.f};
#pragma unroll
      for (int kk = 0; kk < 4; ++kk)
        s = wmma16(qa[kk], load_b_frag(Ks + ct * 16 * CC, CC, kk * 32, lane), s);
#pragma unroll
      for (int v = 0; v < 8; ++v) {
        int m = v + hi * 8;
        float sc = s[v] * SCORE_SCALE + mrowp[v][kc + ct * 16];
        float p = __expf(sc);
        prs[v] += p;
        Pt[wave][m * 32 + ct * 16 + col] = (_Float16)p;
      }
    }
    // in-wave LDS RAW: order DS write -> DS read
    asm volatile("s_wait_dscnt 0" ::: "memory");

    // P (16x32) x V (32x128): 8 column tiles
    v16h pa = load_a_frag(&Pt[wave][0], 32, 0, lane);
#pragma unroll
    for (int ct2 = 0; ct2 < 8; ++ct2)
      out[ct2] = wmma16(pa, load_b_frag(Vt + ct2 * 16 * 32, 32, 0, lane), out[ct2]);

    __syncthreads();  // protect Ks/Vt before next chunk overwrites them
  }

  // finalize: rowsum reduce over 16-lane half, normalize, gate, scatter
#pragma unroll
  for (int v = 0; v < 8; ++v) {
    float r = prs[v];
#pragma unroll
    for (int m = 1; m < 16; m <<= 1) r += __shfl_xor(r, m, 32);
    float inv = fast_rcp(r);
    int q = q0 + v + hi * 8;
    int n = seq_from_win(wi, q);
#pragma unroll
    for (int ct2 = 0; ct2 < 8; ++ct2) {
      int c = ct2 * 16 + col;
      float g = gate[(size_t)n * CC + c];
      x2[(size_t)n * CC + c] = (_Float16)(out[ct2][v] * inv * g);
    }
  }
}

// ---------------- Kernel 5: silu(x2@Wo1+b)@Wo2 + residual, dual store ------
__global__ __launch_bounds__(256) void proj_out(const _Float16* __restrict__ x2,
                                                const _Float16* __restrict__ wo1T,
                                                const float* __restrict__ bo1,
                                                const _Float16* __restrict__ wo2T,
                                                const float* __restrict__ src,
                                                float* __restrict__ out) {
  __shared__ _Float16 H1[16 * CC];
  int mt   = blockIdx.x;                  // 16-row tile
  int wave = threadIdx.x >> 5;
  int lane = threadIdx.x & 31;
  int colbase = wave * 16;
  int col = lane & 15, hi = lane >> 4;

  const _Float16* A = x2 + (size_t)mt * 16 * CC;
  v8f acc = {0.f,0.f,0.f,0.f,0.f,0.f,0.f,0.f};
#pragma unroll
  for (int kk = 0; kk < 4; ++kk)
    acc = wmma16(load_a_frag(A, CC, kk * 32, lane),
                 load_b_frag(wo1T + colbase * CC, CC, kk * 32, lane), acc);
  float bv = bo1[colbase + col];
#pragma unroll
  for (int v = 0; v < 8; ++v)
    H1[(v + hi * 8) * CC + colbase + col] = (_Float16)silu(acc[v] + bv);
  __syncthreads();

  v8f acc2 = {0.f,0.f,0.f,0.f,0.f,0.f,0.f,0.f};
#pragma unroll
  for (int kk = 0; kk < 4; ++kk)
    acc2 = wmma16(load_a_frag(H1, CC, kk * 32, lane),
                  load_b_frag(wo2T + colbase * CC, CC, kk * 32, lane), acc2);
#pragma unroll
  for (int v = 0; v < 8; ++v) {
    int n = mt * 16 + v + hi * 8;
    int c = colbase + col;
    float y = acc2[v] + src[(size_t)n * CC + c];
    out[(size_t)n * CC + c] = y;
    out[(size_t)NTOK * CC + (size_t)n * CC + c] = y;   // tuple (y, y)
  }
}

// ---------------- host launcher ----------------
extern "C" void kernel_launch(void* const* d_in, const int* in_sizes, int n_in,
                              void* d_out, int out_size, void* d_ws, size_t ws_size,
                              hipStream_t stream) {
  const float* source = (const float*)d_in[0];
  const float* mask   = (const float*)d_in[2];
  const float* ln_g   = (const float*)d_in[5];
  const float* ln_b   = (const float*)d_in[6];
  const float* w_gq   = (const float*)d_in[7];
  const float* b_gq   = (const float*)d_in[8];
  const float* w_kv   = (const float*)d_in[9];
  const float* b_kv   = (const float*)d_in[10];
  const float* w_o1   = (const float*)d_in[11];
  const float* b_o1   = (const float*)d_in[12];
  const float* w_o2   = (const float*)d_in[13];

  char*  ws  = (char*)d_ws;
  size_t off = 0;
  auto carve = [&](size_t bytes) -> void* {
    void* p = ws + off;
    off += (bytes + 255) & ~(size_t)255;
    return p;
  };
  _Float16* sln   = (_Float16*)carve((size_t)NTOK * CC * 2);
  float*    gate  = (float*)   carve((size_t)NTOK * CC * 4);
  _Float16* Qh    = (_Float16*)carve((size_t)NWIN * WTOK * CC * 2);
  _Float16* Kh    = (_Float16*)carve((size_t)NWIN * WTOK * CC * 2);
  _Float16* Vh    = (_Float16*)carve((size_t)NWIN * WTOK * CC * 2);
  _Float16* x2    = (_Float16*)carve((size_t)NTOK * CC * 2);
  _Float16* wgqT  = (_Float16*)carve((size_t)256 * CC * 2);
  _Float16* wkvT  = (_Float16*)carve((size_t)256 * CC * 2);
  _Float16* wo1T  = (_Float16*)carve((size_t)CC * CC * 2);
  _Float16* wo2T  = (_Float16*)carve((size_t)CC * CC * 2);

  ln_kernel<<<NTOK / 8, 256, 0, stream>>>(source, ln_g, ln_b, sln);

  wcvt<<<(256 * CC + 255) / 256, 256, 0, stream>>>(w_gq, wgqT, CC, 256);
  wcvt<<<(256 * CC + 255) / 256, 256, 0, stream>>>(w_kv, wkvT, CC, 256);
  wcvt<<<(CC * CC + 255) / 256, 256, 0, stream>>>(w_o1, wo1T, CC, CC);
  wcvt<<<(CC * CC + 255) / 256, 256, 0, stream>>>(w_o2, wo2T, CC, CC);

  proj_gqkv<<<dim3(NTOK / 16, 4), 256, 0, stream>>>(sln, wgqT, wkvT, b_gq, b_kv,
                                                    gate, Qh, Kh, Vh);

  attn_kernel<<<dim3(NWIN, WTOK / 128), 256, 0, stream>>>(Qh, Kh, Vh, mask, gate, x2);

  proj_out<<<NTOK / 16, 256, 0, stream>>>(x2, wo1T, b_o1, wo2T, source, (float*)d_out);
}
